// PointerDecoder_3822520894105
// MI455X (gfx1250) — matrix operand order repaired and checked by reading, not compiled, not run on hardware
//
#include <hip/hip_runtime.h>
#include <hip/hip_bf16.h>
#include <math.h>

// ---------------------------------------------------------------------------
// Problem constants (match reference: B=32, N=256, D=128)
// ---------------------------------------------------------------------------
#define NB   32          // graphs
#define NN   256         // nodes per graph
#define DD   128         // embed dim
#define TD   384         // 3*D (GRU gates)
#define TT   (NB * NN)   // total nodes = 8192
#define STEPS (NN - 1)   // 255 decode steps

typedef _Float16 v16h __attribute__((ext_vector_type(16)));
typedef _Float16 v8h  __attribute__((ext_vector_type(8)));
typedef float    v8f  __attribute__((ext_vector_type(8)));

__device__ __forceinline__ float fast_tanhf(float x) {
#if defined(__has_builtin) && __has_builtin(__builtin_amdgcn_tanhf)
    return __builtin_amdgcn_tanhf(x);   // v_tanh_f32 on gfx1250
#else
    return tanhf(x);
#endif
}

// ---------------------------------------------------------------------------
// Small prep kernels: f32->f16 convert, f32 transpose
// ---------------------------------------------------------------------------
__global__ void cvt_f16_k(const float* __restrict__ s, _Float16* __restrict__ d, int n) {
    int i = blockIdx.x * blockDim.x + threadIdx.x;
    if (i < n) d[i] = (_Float16)s[i];
}

__global__ void transpose_f32_k(const float* __restrict__ s, float* __restrict__ d,
                                int rows, int cols) {
    int i = blockIdx.x * blockDim.x + threadIdx.x;
    if (i < rows * cols) {
        int r = i / cols, c = i - r * cols;
        d[(size_t)c * rows + r] = s[i];
    }
}

// ---------------------------------------------------------------------------
// WMMA GEMM:  C[M,N] = A[M,K] (f16, row-major) x W[N,K]^T (f16, row-major) + bias
// mode 0: store f32            (GI_all)
// mode 1: relu, store f16      (hidden of key MLP)
// mode 2: l2norm rows, f32     (keys; requires blockIdx.y==0 covering all N=128)
// One WG = 8 waves; wave w computes the 16x16 tile at cols [w*16, w*16+16).
// A and B both use contiguous k = 16*hi + s order for v_wmma_f32_16x16x32_f16.
// ---------------------------------------------------------------------------
__global__ __launch_bounds__(256)
void wmma_gemm_k(const _Float16* __restrict__ A, const _Float16* __restrict__ W,
                 const float* __restrict__ bias,
                 float* __restrict__ Cf, _Float16* __restrict__ Ch,
                 int M, int N, int K, int mode) {
    const int lane = threadIdx.x & 31;
    const int wave = threadIdx.x >> 5;
    const int hi   = lane >> 4;                    // 0: lanes 0-15, 1: lanes 16-31
    const int mrow = blockIdx.x * 16;
    const int ncol = blockIdx.y * 128 + wave * 16;
    const int m    = mrow + (lane & 15);
    const int n    = ncol + (lane & 15);
    const int hh   = hi * 8;                       // unused; A and B both offset by koff
    const int koff = hi * 16;                      // B: K-offset split 0..15 / 16..31

    v8f c = {0.f, 0.f, 0.f, 0.f, 0.f, 0.f, 0.f, 0.f};
    for (int kb = 0; kb < K; kb += 32) {
        union { v16h v; v8h h[2]; } a;
        const _Float16* pa = A + (size_t)m * K + kb;
        a.h[0] = *reinterpret_cast<const v8h*>(pa + koff);      // K = kb+koff .. +7
        a.h[1] = *reinterpret_cast<const v8h*>(pa + koff + 8);  // K = kb+koff+8 .. +15
        v16h b = *reinterpret_cast<const v16h*>(W + (size_t)n * K + kb + koff);
        c = __builtin_amdgcn_wmma_f32_16x16x32_f16(false, a.v, false, b,
                                                   (short)0, c, false, false);
    }

    const float bn = bias ? bias[n] : 0.f;
    if (mode == 1) {
#pragma unroll
        for (int r = 0; r < 8; ++r) {
            int row = mrow + r + hi * 8;
            float x = c[r] + bn;
            Ch[(size_t)row * N + n] = (_Float16)(x > 0.f ? x : 0.f);
        }
    } else if (mode == 2) {
        __shared__ float rowss[16];
        if (threadIdx.x < 16) rowss[threadIdx.x] = 0.f;
        __syncthreads();
        float vals[8];
#pragma unroll
        for (int r = 0; r < 8; ++r) {
            vals[r] = c[r] + bn;
            atomicAdd(&rowss[r + hi * 8], vals[r] * vals[r]);   // ds_add_f32
        }
        __syncthreads();
#pragma unroll
        for (int r = 0; r < 8; ++r) {
            int rl = r + hi * 8;
            float nrm = fmaxf(sqrtf(rowss[rl]), 1e-12f);
            Cf[(size_t)(mrow + rl) * N + n] = vals[r] / nrm;
        }
    } else {
#pragma unroll
        for (int r = 0; r < 8; ++r) {
            int row = mrow + r + hi * 8;
            Cf[(size_t)row * N + n] = c[r] + bn;
        }
    }
}

// ---------------------------------------------------------------------------
// Persistent decode kernel: one workgroup per graph, 255 sequential steps.
// Dynamic LDS layout (floats):
// ---------------------------------------------------------------------------
#define KTP        257                        // keysT pitch (conflict-free)
#define OFF_KEYST  0                          // [128][257]
#define OFF_GH     (OFF_KEYST + DD * KTP)     // 384
#define OFF_H      (OFF_GH + TD)              // 128
#define OFF_T1     (OFF_H + DD)               // 128 (also gctx in prologue)
#define OFF_Q      (OFF_T1 + DD)              // 128
#define OFF_V      (OFF_Q + DD)               // 128
#define OFF_BHH    (OFF_V + DD)               // 384
#define OFF_BQ1    (OFF_BHH + TD)             // 128
#define OFF_BQ2    (OFF_BQ1 + DD)             // 128
#define OFF_VIS    (OFF_BQ2 + DD)             // 256 ints
#define OFF_WRED   (OFF_VIS + NN)             // 8
#define OFF_WREDI  (OFF_WRED + 8)             // 8 ints
#define OFF_SCAL   (OFF_WREDI + 8)            // 8 slots
#define SMEM_FLOATS (OFF_SCAL + 8)
#define SMEM_BYTES  (SMEM_FLOATS * 4)

__device__ __forceinline__ float wave_sum(float v) {
#pragma unroll
    for (int off = 16; off > 0; off >>= 1) v += __shfl_down(v, off, 32);
    return v;
}
__device__ __forceinline__ void wave_argmax(float& v, int& i) {
#pragma unroll
    for (int off = 16; off > 0; off >>= 1) {
        float ov = __shfl_down(v, off, 32);
        int   oi = __shfl_down(i, off, 32);
        if (ov > v || (ov == v && oi < i)) { v = ov; i = oi; }
    }
}

__global__ __launch_bounds__(256)
void decode_k(const float* __restrict__ E,        // [T,D] node embeddings
              const int*   __restrict__ start_nodes,
              const float* __restrict__ keysg,    // [T,D] normalized keys
              const float* __restrict__ GI,       // [T,3D] E@W_ih^T + b_ih
              const float* __restrict__ WhhT,     // [D,3D]
              const float* __restrict__ Wq1T,     // [D,D]
              const float* __restrict__ Wq2T,     // [D,D]
              const float* __restrict__ WhT,      // [D,D]
              const float* __restrict__ b_hh,     // [3D]
              const float* __restrict__ bq1, const float* __restrict__ bq2,
              const float* __restrict__ bhg,      // bh [D]
              const float* __restrict__ vvec,     // [D]
              float* __restrict__ out_tours,      // [B,256]
              float* __restrict__ out_lp) {       // [B,255]
    extern __shared__ float smem[];
    float* keysT = smem + OFF_KEYST;
    float* gh    = smem + OFF_GH;
    float* hvec  = smem + OFF_H;
    float* t1    = smem + OFF_T1;
    float* qv    = smem + OFF_Q;
    float* vs    = smem + OFF_V;
    float* bhh   = smem + OFF_BHH;
    float* bq1s  = smem + OFF_BQ1;
    float* bq2s  = smem + OFF_BQ2;
    int*   vis   = (int*)(smem + OFF_VIS);
    float* wred  = smem + OFF_WRED;
    int*   wredi = (int*)(smem + OFF_WREDI);
    float* scal  = smem + OFF_SCAL;
    int*   scali = (int*)scal;

    const int tid  = threadIdx.x;
    const int lane = tid & 31;
    const int wave = tid >> 5;
    const int b    = blockIdx.x;
    const int base = b * NN;

    // ---- stage keys (transposed) into LDS; coalesced global, conflict-free LDS
    for (int idx = tid; idx < NN * DD; idx += 256) {
        int t = idx >> 7, d = idx & 127;
        keysT[d * KTP + t] = keysg[(size_t)(base + t) * DD + d];
    }
    if (tid < DD) { vs[tid] = vvec[tid]; bq1s[tid] = bq1[tid]; bq2s[tid] = bq2[tid]; }
    for (int i = tid; i < TD; i += 256) bhh[i] = b_hh[i];
    vis[tid] = 0;

    // ---- gctx = mean of graph embeddings; hidden0 = gctx@Wh^T + bh
    if (tid < DD) {
        float acc = 0.f;
        for (int t = 0; t < NN; ++t) acc += E[(size_t)(base + t) * DD + tid];
        t1[tid] = acc * (1.f / NN);
    }
    __syncthreads();
    if (tid < DD) {
        float acc = bhg[tid];
#pragma unroll 4
        for (int k = 0; k < DD; ++k) acc = fmaf(WhT[(size_t)k * DD + tid], t1[k], acc);
        hvec[tid] = acc;
    }
    if (tid == 0) scali[0] = start_nodes[b] - base;   // local current node

    // ---- sequential greedy decode
    for (int step = 0; step < STEPS; ++step) {
        __syncthreads();
        const int cur = scali[0];
        if (tid == 0) {
            vis[cur] = 1;
            out_tours[(size_t)b * NN + step] = (float)(base + cur);
        }
        // gh = h @ W_hh^T + b_hh   (384 rows over 256 threads)
        {
            float acc = bhh[tid];
#pragma unroll 4
            for (int k = 0; k < DD; ++k)
                acc = fmaf(WhhT[(size_t)k * TD + tid], hvec[k], acc);
            gh[tid] = acc;
            if (tid < DD) {
                float acc1 = bhh[tid + 256];
#pragma unroll 4
                for (int k = 0; k < DD; ++k)
                    acc1 = fmaf(WhhT[(size_t)k * TD + tid + 256], hvec[k], acc1);
                gh[tid + 256] = acc1;
            }
        }
        __syncthreads();
        // GRU combine (gate order r,z,n); gi from precomputed GI_all
        if (tid < DD) {
            const size_t g0 = (size_t)(base + cur) * TD;
            float gir = GI[g0 + tid], giz = GI[g0 + DD + tid], gin = GI[g0 + 2 * DD + tid];
            float r = 1.f / (1.f + expf(-(gir + gh[tid])));
            float z = 1.f / (1.f + expf(-(giz + gh[DD + tid])));
            float n = fast_tanhf(gin + r * gh[2 * DD + tid]);
            hvec[tid] = (1.f - z) * n + z * hvec[tid];
        }
        __syncthreads();
        // t1 = relu(h @ Wq1^T + bq1)
        if (tid < DD) {
            float acc = bq1s[tid];
#pragma unroll 4
            for (int k = 0; k < DD; ++k) acc = fmaf(Wq1T[(size_t)k * DD + tid], hvec[k], acc);
            t1[tid] = acc > 0.f ? acc : 0.f;
        }
        __syncthreads();
        // q = l2norm(t1 @ Wq2^T + bq2)
        float qacc = 0.f;
        if (tid < DD) {
            qacc = bq2s[tid];
#pragma unroll 4
            for (int k = 0; k < DD; ++k) qacc = fmaf(Wq2T[(size_t)k * DD + tid], t1[k], qacc);
            qv[tid] = qacc;
        }
        {
            float sq = (tid < DD) ? qacc * qacc : 0.f;
            float ps = wave_sum(sq);
            if (lane == 0) wred[wave] = ps;
        }
        __syncthreads();
        if (tid == 0) {
            float s = 0.f;
            for (int i = 0; i < 8; ++i) s += wred[i];
            scal[3] = 1.f / fmaxf(sqrtf(s), 1e-12f);
        }
        __syncthreads();
        if (tid < DD) qv[tid] *= scal[3];
        __syncthreads();
        // scores: one node per thread; tanh from LDS, keys conflict-free
        float sc = 0.f;
#pragma unroll 4
        for (int d = 0; d < DD; ++d)
            sc = fmaf(vs[d], fast_tanhf(keysT[d * KTP + tid] + qv[d]), sc);
        float sval = vis[tid] ? -INFINITY : sc;
        int   sidx = tid;
        wave_argmax(sval, sidx);
        if (lane == 0) { wred[wave] = sval; wredi[wave] = sidx; }
        __syncthreads();
        if (tid == 0) {
            float bv = wred[0]; int bi = wredi[0];
            for (int i = 1; i < 8; ++i)
                if (wred[i] > bv || (wred[i] == bv && wredi[i] < bi)) { bv = wred[i]; bi = wredi[i]; }
            scal[1] = bv; scali[2] = bi;
        }
        __syncthreads();
        const float smax = scal[1];
        float e = vis[tid] ? 0.f : expf(sc - smax);
        float pe = wave_sum(e);
        if (lane == 0) wred[wave] = pe;
        __syncthreads();
        if (tid == 0) {
            float Z = 0.f;
            for (int i = 0; i < 8; ++i) Z += wred[i];
            out_lp[(size_t)b * STEPS + step] = logf(1.f / Z + 1e-10f);
            scali[0] = scali[2];                // advance to argmax node
        }
    }
    __syncthreads();
    if (tid == 0) out_tours[(size_t)b * NN + (NN - 1)] = (float)(base + scali[0]);
}

// ---------------------------------------------------------------------------
// Host entry
// ---------------------------------------------------------------------------
extern "C" void kernel_launch(void* const* d_in, const int* in_sizes, int n_in,
                              void* d_out, int out_size, void* d_ws, size_t ws_size,
                              hipStream_t stream) {
    const float* E    = (const float*)d_in[0];
    const int*   strt = (const int*)d_in[1];
    // d_in[2] = batch_idx (contiguous blocks; implied by layout)
    const float* Wq1  = (const float*)d_in[3];
    const float* bq1  = (const float*)d_in[4];
    const float* Wq2  = (const float*)d_in[5];
    const float* bq2  = (const float*)d_in[6];
    const float* Wk1  = (const float*)d_in[7];
    const float* bk1  = (const float*)d_in[8];
    const float* Wk2  = (const float*)d_in[9];
    const float* bk2  = (const float*)d_in[10];
    const float* W_ih = (const float*)d_in[11];
    const float* W_hh = (const float*)d_in[12];
    const float* b_ih = (const float*)d_in[13];
    const float* b_hh = (const float*)d_in[14];
    const float* v    = (const float*)d_in[15];
    const float* Wh   = (const float*)d_in[16];
    const float* bh   = (const float*)d_in[17];

    float* out_tours = (float*)d_out;           // [32*256]
    float* out_lp    = (float*)d_out + NB * NN; // [32*255]

    // workspace carve (256B aligned)
    char* ws = (char*)d_ws;
    auto carve = [&](size_t bytes) { char* p = ws; ws += (bytes + 255) & ~(size_t)255; return p; };
    _Float16* E16   = (_Float16*)carve((size_t)TT * DD * 2);
    _Float16* H1    = (_Float16*)carve((size_t)TT * DD * 2);
    _Float16* Wk1h  = (_Float16*)carve((size_t)DD * DD * 2);
    _Float16* Wk2h  = (_Float16*)carve((size_t)DD * DD * 2);
    _Float16* Wihh  = (_Float16*)carve((size_t)TD * DD * 2);
    float*    keys  = (float*)carve((size_t)TT * DD * 4);
    float*    GI    = (float*)carve((size_t)TT * TD * 4);
    float*    WhhT  = (float*)carve((size_t)DD * TD * 4);
    float*    Wq1T  = (float*)carve((size_t)DD * DD * 4);
    float*    Wq2T  = (float*)carve((size_t)DD * DD * 4);
    float*    WhT   = (float*)carve((size_t)DD * DD * 4);

    // prep: f16 conversions + weight transposes
    cvt_f16_k<<<(TT * DD + 255) / 256, 256, 0, stream>>>(E, E16, TT * DD);
    cvt_f16_k<<<(DD * DD + 255) / 256, 256, 0, stream>>>(Wk1, Wk1h, DD * DD);
    cvt_f16_k<<<(DD * DD + 255) / 256, 256, 0, stream>>>(Wk2, Wk2h, DD * DD);
    cvt_f16_k<<<(TD * DD + 255) / 256, 256, 0, stream>>>(W_ih, Wihh, TD * DD);
    transpose_f32_k<<<(TD * DD + 255) / 256, 256, 0, stream>>>(W_hh, WhhT, TD, DD);
    transpose_f32_k<<<(DD * DD + 255) / 256, 256, 0, stream>>>(Wq1, Wq1T, DD, DD);
    transpose_f32_k<<<(DD * DD + 255) / 256, 256, 0, stream>>>(Wq2, Wq2T, DD, DD);
    transpose_f32_k<<<(DD * DD + 255) / 256, 256, 0, stream>>>(Wh, WhT, DD, DD);

    // WMMA precompute: key MLP (+l2norm) and GI_all = E@W_ih^T + b_ih
    wmma_gemm_k<<<dim3(TT / 16, 1), 256, 0, stream>>>(E16, Wk1h, bk1, nullptr, H1,
                                                      TT, DD, DD, 1);
    wmma_gemm_k<<<dim3(TT / 16, 1), 256, 0, stream>>>(H1, Wk2h, bk2, keys, nullptr,
                                                      TT, DD, DD, 2);
    wmma_gemm_k<<<dim3(TT / 16, 3), 256, 0, stream>>>(E16, Wihh, b_ih, GI, nullptr,
                                                      TT, TD, DD, 0);

    // persistent per-graph decode (dynamic LDS: keys + state, ~135.6 KB/WG)
    decode_k<<<NB, 256, SMEM_BYTES, stream>>>(E, strt, keys, GI, WhhT, Wq1T, Wq2T,
                                              WhT, b_hh, bq1, bq2, bh, v,
                                              out_tours, out_lp);
}
